// LayerNormMLP_4870492913837
// MI455X (gfx1250) — matrix operand order repaired and checked by reading, not compiled, not run on hardware
//
#include <hip/hip_runtime.h>
#include <hip/hip_bf16.h>
#include <math.h>

typedef __attribute__((ext_vector_type(16))) __bf16 v16bf;
typedef __attribute__((ext_vector_type(8)))  __bf16 v8bf;
typedef __attribute__((ext_vector_type(8)))  float  v8f;
typedef __attribute__((ext_vector_type(4)))  int    v4i;

#define TOKENS 8192
#define HIDDEN 4096
#define FFN    16384

#define AS1 __attribute__((address_space(1)))
#define AS3 __attribute__((address_space(3)))

#if __has_builtin(__builtin_amdgcn_global_load_async_to_lds_b128)
#define USE_ASYNC_LDS 1
#endif

// 16-byte global -> LDS copy; async (ASYNCcnt) when available
static __device__ __forceinline__ void async_copy16(const __bf16* g, __bf16* l) {
#if defined(USE_ASYNC_LDS)
    __builtin_amdgcn_global_load_async_to_lds_b128(
        (AS1 v4i*)(v4i*)(void*)const_cast<__bf16*>(g),
        (AS3 v4i*)(v4i*)(void*)l, 0, 0);
#else
    *(v4i*)l = *(const v4i*)g;
#endif
}

static __device__ __forceinline__ void wait_async_all() {
#if defined(USE_ASYNC_LDS)
#if __has_builtin(__builtin_amdgcn_s_wait_asynccnt)
    __builtin_amdgcn_s_wait_asynccnt(0);
#else
    asm volatile("s_wait_asynccnt 0" ::: "memory");
#endif
#endif
}

// cheap branch-free GELU (tanh form via sigmoid): x * sigmoid(1.5958x + 0.07135x^3)
static __device__ __forceinline__ float gelu_fast(float x) {
    float z = x * (1.59576912f + 0.071354816f * x * x);
    float e = __expf(-z);                       // v_exp_f32
    return x * __builtin_amdgcn_rcpf(1.0f + e); // v_rcp_f32
}

// ---------------------------------------------------------------------------
// Kernel 1: LayerNorm over hidden dim, fp32 in -> bf16 out
// ---------------------------------------------------------------------------
__global__ void __launch_bounds__(256) ln_bf16_kernel(
    const float* __restrict__ x, const float* __restrict__ gamma,
    const float* __restrict__ beta, __bf16* __restrict__ out)
{
    const int row = blockIdx.x;
    const int t   = threadIdx.x;
    const float* xr = x + (size_t)row * HIDDEN;

    float4 v[4];
#pragma unroll
    for (int i = 0; i < 4; ++i) v[i] = ((const float4*)xr)[t * 4 + i];

    float s = 0.f, s2 = 0.f;
#pragma unroll
    for (int i = 0; i < 4; ++i) {
        s  += v[i].x + v[i].y + v[i].z + v[i].w;
        s2 += v[i].x*v[i].x + v[i].y*v[i].y + v[i].z*v[i].z + v[i].w*v[i].w;
    }
#pragma unroll
    for (int off = 16; off > 0; off >>= 1) {
        s  += __shfl_xor(s,  off, 32);
        s2 += __shfl_xor(s2, off, 32);
    }
    __shared__ float ws1[8], ws2[8];
    const int wave = t >> 5, lane = t & 31;
    if (lane == 0) { ws1[wave] = s; ws2[wave] = s2; }
    __syncthreads();
    if (t == 0) {
        float a = 0.f, b = 0.f;
#pragma unroll
        for (int i = 0; i < 8; ++i) { a += ws1[i]; b += ws2[i]; }
        ws1[0] = a; ws2[0] = b;
    }
    __syncthreads();
    const float mu  = ws1[0] * (1.0f / HIDDEN);
    const float var = ws2[0] * (1.0f / HIDDEN) - mu * mu;
    const float rs  = rsqrtf(var + 1e-5f);

    float4 g[4], b4[4];
#pragma unroll
    for (int i = 0; i < 4; ++i) {
        g[i]  = ((const float4*)gamma)[t * 4 + i];
        b4[i] = ((const float4*)beta)[t * 4 + i];
    }
    v16bf o;
#pragma unroll
    for (int i = 0; i < 4; ++i) {
        const float* fv = (const float*)&v[i];
        const float* fg = (const float*)&g[i];
        const float* fb = (const float*)&b4[i];
#pragma unroll
        for (int j = 0; j < 4; ++j)
            o[i * 4 + j] = (__bf16)((fv[j] - mu) * rs * fg[j] + fb[j]);
    }
    *(v16bf*)&out[(size_t)row * HIDDEN + t * 16] = o;
}

// ---------------------------------------------------------------------------
// GEMM: C = act(A_bf16[M,K] * B_f32->bf16[K,N] + bias)
// Block 256x256x32, 512 thr = 16 waves (4x4), wave tile 64x64 -> 16 wmma/K.
// Double-buffered LDS; A staged with async global->LDS copies (ASYNCcnt),
// B register-pipelined one K-tile ahead with in-register fp32->bf16 cvt.
// blockIdx.x walks M so concurrent blocks share the same weight panel in L2.
// ---------------------------------------------------------------------------
template <typename OutT, bool GELU>
__global__ void __launch_bounds__(512, 1) gemm_bf16_kernel(
    const __bf16* __restrict__ A, const float* __restrict__ B,
    const float* __restrict__ bias, OutT* __restrict__ C,
    int M, int N, int K)
{
    constexpr int BM = 256, BN = 256, BK = 32;
    constexpr int ASd = BK + 8;    // padded LDS strides (conflict-free)
    constexpr int BSd = BN + 8;
    __shared__ __bf16 As[2][BM * ASd];
    __shared__ __bf16 Bs[2][BK * BSd];

    const int t     = threadIdx.x;
    const int lane  = t & 31;
    const int wave  = t >> 5;            // 0..15
    const int waveM = wave & 3;          // 4 waves along M (64 rows each)
    const int waveN = wave >> 2;         // 4 waves along N (64 cols each)
    const int m0    = blockIdx.x * BM;   // x walks M -> weight-panel L2 reuse
    const int n0    = blockIdx.y * BN;
    const int row16 = lane & 15;
    const int kg    = lane >> 4;

    // staging coords (512 threads)
    const int arow = t >> 1, acg = (t & 1) * 16;    // A: 256 rows x 2 chunks
    const int brow = t >> 4, bcg = (t & 15) * 16;   // B: 32 rows x 16 chunks

    const __bf16* Ag = A + (size_t)(m0 + arow) * K + acg;
    const float*  Bg = B + (size_t)brow * N + n0 + bcg;

    const v8f zacc = {};
    v8f acc[4][4];
#pragma unroll
    for (int mt = 0; mt < 4; ++mt)
#pragma unroll
        for (int nt = 0; nt < 4; ++nt) acc[mt][nt] = zacc;

    float4 br[4];                                   // B pipeline registers

    auto loadB = [&]() {
#pragma unroll
        for (int i = 0; i < 4; ++i) br[i] = ((const float4*)Bg)[i];
        __builtin_prefetch(Bg + (size_t)(2 * BK) * N, 0, 1);  // 2 tiles ahead
        Bg += (size_t)BK * N;
    };
    auto storeB = [&](int buf) {
        v16bf bb;
#pragma unroll
        for (int i = 0; i < 4; ++i) {
            const float* f = (const float*)&br[i];
            bb[i*4+0] = (__bf16)f[0]; bb[i*4+1] = (__bf16)f[1];
            bb[i*4+2] = (__bf16)f[2]; bb[i*4+3] = (__bf16)f[3];
        }
        *(v16bf*)&Bs[buf][brow * BSd + bcg] = bb;
    };
    auto stageA = [&](int buf) {
        __bf16* dst = &As[buf][arow * ASd + acg];
        async_copy16(Ag,     dst);
        async_copy16(Ag + 8, dst + 8);
        Ag += BK;
    };

    loadB();
    stageA(0);

    int buf = 0;
    for (int k0 = 0; k0 < K; k0 += BK) {
        storeB(buf);
        wait_async_all();
        __syncthreads();                 // tiles in [buf] ready

        if (k0 + BK < K) {               // stage next tile into other buffer
            loadB();
            stageA(buf ^ 1);
        }

        // B fragments: lane = N column, 16 K per lane (table order, matches A)
        v16bf bfr[4];
#pragma unroll
        for (int nt = 0; nt < 4; ++nt)
#pragma unroll
            for (int s = 0; s < 16; ++s)
                bfr[nt][s] = Bs[buf][((s < 8 ? s : s + 8) + kg * 8) * BSd + waveN * 64 + nt * 16 + row16];

#pragma unroll
        for (int mt = 0; mt < 4; ++mt) {
            const __bf16* base = &As[buf][(waveM * 64 + mt * 16 + row16) * ASd];
            v8bf lo = *(const v8bf*)(base + kg * 8);
            v8bf hi = *(const v8bf*)(base + 16 + kg * 8);
            v16bf af;
#pragma unroll
            for (int i = 0; i < 8; ++i) { af[i] = lo[i]; af[8 + i] = hi[i]; }
#pragma unroll
            for (int nt = 0; nt < 4; ++nt)
                acc[mt][nt] = __builtin_amdgcn_wmma_f32_16x16x32_bf16(
                    false, af, false, bfr[nt],
                    (short)0, acc[mt][nt], false, false);
        }
        buf ^= 1;
        __syncthreads();                 // computed before buffers reused
    }

    // epilogue: C/D layout -> N = lane&15, M = (lane>>4)*8 + vgpr
    const int nlane = lane & 15;
    const int mbase = (lane >> 4) * 8;
#pragma unroll
    for (int nt = 0; nt < 4; ++nt) {
        const int ncol = n0 + waveN * 64 + nt * 16 + nlane;
        const float bs = bias[ncol];
#pragma unroll
        for (int mt = 0; mt < 4; ++mt) {
#pragma unroll
            for (int i = 0; i < 8; ++i) {
                const int mrow = m0 + waveM * 64 + mt * 16 + mbase + i;
                float val = acc[mt][nt][i] + bs;
                if (GELU) val = gelu_fast(val);
                C[(size_t)mrow * N + ncol] = (OutT)val;
            }
        }
    }
}

// ---------------------------------------------------------------------------
extern "C" void kernel_launch(void* const* d_in, const int* in_sizes, int n_in,
                              void* d_out, int out_size, void* d_ws, size_t ws_size,
                              hipStream_t stream) {
    (void)in_sizes; (void)n_in; (void)out_size; (void)ws_size;
    const float* x     = (const float*)d_in[0];
    const float* gam   = (const float*)d_in[1];
    const float* bet   = (const float*)d_in[2];
    const float* fc1_k = (const float*)d_in[3];
    const float* fc1_b = (const float*)d_in[4];
    const float* fc2_k = (const float*)d_in[5];
    const float* fc2_b = (const float*)d_in[6];
    float* out = (float*)d_out;

    __bf16* ln = (__bf16*)d_ws;                       // 8192x4096  bf16 (64 MB)
    __bf16* h  = ln + (size_t)TOKENS * HIDDEN;        // 8192x16384 bf16 (256 MB)

    ln_bf16_kernel<<<TOKENS, 256, 0, stream>>>(x, gam, bet, ln);

    gemm_bf16_kernel<__bf16, true>
        <<<dim3(TOKENS / 256, FFN / 256), 512, 0, stream>>>(
            ln, fc1_k, fc1_b, h, TOKENS, FFN, HIDDEN);

    gemm_bf16_kernel<float, false>
        <<<dim3(TOKENS / 256, HIDDEN / 256), 512, 0, stream>>>(
            h, fc2_k, fc2_b, out, TOKENS, HIDDEN, FFN);
}
